// HybridMamba2Block_49074296324243
// MI455X (gfx1250) — matrix operand order repaired and checked by reading, not compiled, not run on hardware
//
#include <hip/hip_runtime.h>
#include <hip/hip_bf16.h>
#include <stdint.h>

// ---------------- problem constants ----------------
#define BSZ 2
#define TT 2048
#define DD 1024
#define HATT 16
#define HDIM 64
#define WINSZ 128
#define HS 8
#define DI 2048
#define NS 16
#define PPC 256          // P = DI/HS
#define DTR 64
#define LDP 4480         // padded TOT (2*DI + DTR + 2*HS*NS = 4416 -> 4480)
#define RTOT (BSZ*TT)    // 4096 rows

typedef unsigned short bfbits;
typedef __attribute__((ext_vector_type(16))) __bf16 v16bf;
typedef __attribute__((ext_vector_type(8)))  float  v8f;
typedef __attribute__((ext_vector_type(4)))  unsigned int u32x4;

union Frag { v16bf v; u32x4 q[2]; };

__device__ inline bfbits f2bf(float f){
  union { float f; uint32_t u; } x; x.f = f;
  return (bfbits)((x.u + 0x7FFFu + ((x.u >> 16) & 1u)) >> 16);   // RNE
}
__device__ inline float sigm(float x){ return 1.f/(1.f+__expf(-x)); }

__device__ inline float blockReduceSum(float v){
  #pragma unroll
  for(int m=16;m>=1;m>>=1) v += __shfl_xor(v, m);
  __shared__ float red[8];
  int lane=threadIdx.x&31, w=threadIdx.x>>5;
  if(lane==0) red[w]=v;
  __syncthreads();
  float t = (threadIdx.x < 8) ? red[threadIdx.x] : 0.f;
  #pragma unroll
  for(int m=4;m>=1;m>>=1) t += __shfl_xor(t, m);
  if(threadIdx.x==0) red[0]=t;
  __syncthreads();
  float r = red[0];
  __syncthreads();
  return r;
}

// ---------------- generic bf16 WMMA GEMM:  C[M,N] = A[M,K] * W[N,K]^T (+ res) ----------------
// grid = (N/128, M/128), block = 256 (8 waves). Double-buffered LDS 128x128x32 tiles:
// global loads for tile k+1 are issued before the WMMAs of tile k, so HBM/L2 latency
// hides under 64 WMMAs per block per K-step; one barrier per K-tile.
__global__ __launch_bounds__(256)
void gemm_bf16_wmma(const bfbits* __restrict__ A, int lda,
                    const bfbits* __restrict__ W, int ldw,
                    float* __restrict__ C, int ldc,
                    const float* __restrict__ res, int ldres,
                    int K)
{
  __shared__ __align__(16) bfbits As[2][128*40];   // 40-half row stride (80B, 16B multiple)
  __shared__ __align__(16) bfbits Bs[2][128*40];
  const int tid=threadIdx.x, lane=tid&31, wid=tid>>5;
  const int lrow=lane&15, lhi=lane>>4;
  const int waveM=wid&1, waveN=wid>>1;             // 2 x 4 wave grid -> 64x32 per wave
  const size_t am = (size_t)blockIdx.y*128;
  const size_t wn = (size_t)blockIdx.x*128;

  // per-thread staging coordinates: two 16B chunks per matrix per tile
  const int row0 = tid>>2, quad = tid&3;
  const int row1 = row0 + 64;
  const bfbits* Ap0 = A + (am+row0)*lda + quad*8;
  const bfbits* Ap1 = A + (am+row1)*lda + quad*8;
  const bfbits* Wp0 = W + (wn+row0)*ldw + quad*8;
  const bfbits* Wp1 = W + (wn+row1)*ldw + quad*8;
  const int so0 = row0*40 + quad*8;
  const int so1 = row1*40 + quad*8;

  v8f acc[4][2];
  #pragma unroll
  for(int mi=0;mi<4;++mi){
    #pragma unroll
    for(int ni=0;ni<2;++ni) acc[mi][ni] = (v8f){};
  }

  const int ktiles = K>>5;

  // prologue: stage tile 0
  u32x4 ra0 = *(const u32x4*)Ap0;
  u32x4 ra1 = *(const u32x4*)Ap1;
  u32x4 rb0 = *(const u32x4*)Wp0;
  u32x4 rb1 = *(const u32x4*)Wp1;
  *(u32x4*)&As[0][so0] = ra0;
  *(u32x4*)&As[0][so1] = ra1;
  *(u32x4*)&Bs[0][so0] = rb0;
  *(u32x4*)&Bs[0][so1] = rb1;
  __syncthreads();

  for(int kt=0; kt<ktiles; ++kt){
    const int cur = kt & 1;
    const bool more = (kt+1) < ktiles;
    if(more){                       // issue next-tile global loads (latency hidden by WMMAs)
      const size_t ko = (size_t)(kt+1)*32;
      ra0 = *(const u32x4*)(Ap0 + ko);
      ra1 = *(const u32x4*)(Ap1 + ko);
      rb0 = *(const u32x4*)(Wp0 + ko);
      rb1 = *(const u32x4*)(Wp1 + ko);
      if(kt+2 < ktiles){            // -> global_prefetch_b8 two tiles ahead
        __builtin_prefetch(Ap0 + ko + 32, 0, 1);
        __builtin_prefetch(Wp0 + ko + 32, 0, 1);
      }
    }

    Frag a[4];
    #pragma unroll
    for(int mi=0;mi<4;++mi){
      const bfbits* pa = &As[cur][(waveM*64 + mi*16 + lrow)*40 + lhi*8];
      a[mi].q[0] = *(const u32x4*)pa;
      a[mi].q[1] = *(const u32x4*)(pa + 16);
    }
    #pragma unroll
    for(int ni=0;ni<2;++ni){
      Frag b;
      const bfbits* pb = &Bs[cur][(waveN*32 + ni*16 + lrow)*40 + lhi*8];
      b.q[0] = *(const u32x4*)pb;
      b.q[1] = *(const u32x4*)(pb + 16);
      #pragma unroll
      for(int mi=0;mi<4;++mi)
        acc[mi][ni] = __builtin_amdgcn_wmma_f32_16x16x32_bf16(
            false, a[mi].v, false, b.v, (short)0, acc[mi][ni], false, false);
    }

    if(more){                       // stage tile k+1 into the alternate buffer
      *(u32x4*)&As[cur^1][so0] = ra0;
      *(u32x4*)&As[cur^1][so1] = ra1;
      *(u32x4*)&Bs[cur^1][so0] = rb0;
      *(u32x4*)&Bs[cur^1][so1] = rb1;
    }
    __syncthreads();
  }

  #pragma unroll
  for(int mi=0;mi<4;++mi){
    #pragma unroll
    for(int ni=0;ni<2;++ni){
      #pragma unroll
      for(int v=0;v<8;++v){
        size_t gr = am + waveM*64 + mi*16 + lhi*8 + v;
        size_t gc = wn + waveN*32 + ni*16 + lrow;
        float val = acc[mi][ni][v];
        if(res) val += res[gr*(size_t)ldres + gc];
        C[gr*(size_t)ldc + gc] = val;
      }
    }
  }
}

// ---------------- elementwise / small kernels ----------------
__global__ void cast_pad_bf16(bfbits* __restrict__ dst, const float* __restrict__ src,
                              long nsrc, long ndst){
  long i = (long)blockIdx.x*256 + threadIdx.x;
  if(i < ndst) dst[i] = (i < nsrc) ? f2bf(src[i]) : (bfbits)0;
}

__global__ void cast_f32_bf16(bfbits* __restrict__ dst, const float* __restrict__ src, long n){
  long i = (long)blockIdx.x*256 + threadIdx.x;
  if(i < n) dst[i] = f2bf(src[i]);
}

__global__ __launch_bounds__(256)
void rmsnorm_bf16(const float* __restrict__ x, const float* __restrict__ w,
                  bfbits* __restrict__ out, int Dl){
  int r = blockIdx.x, tid = threadIdx.x;
  int per = Dl >> 8;
  float v[8]; float ss = 0.f;
  for(int i=0;i<per;++i){ v[i] = x[(size_t)r*Dl + i*256 + tid]; ss += v[i]*v[i]; }
  float tot = blockReduceSum(ss);
  float rs = rsqrtf(tot/(float)Dl + 1e-6f);
  for(int i=0;i<per;++i)
    out[(size_t)r*Dl + i*256 + tid] = f2bf(v[i]*rs*w[i*256 + tid]);
}

__global__ __launch_bounds__(256)
void gate_rms_bf16(const float* __restrict__ y, const float* __restrict__ proj,
                   const float* __restrict__ w, bfbits* __restrict__ out){
  int r = blockIdx.x, tid = threadIdx.x;
  float v[8]; float ss = 0.f;
  #pragma unroll
  for(int i=0;i<8;++i){
    int c = i*256 + tid;
    float z = proj[(size_t)r*LDP + DI + c];
    float t = y[(size_t)r*DI + c] * z * sigm(z);
    v[i] = t; ss += t*t;
  }
  float tot = blockReduceSum(ss);
  float rs = rsqrtf(tot/(float)DI + 1e-6f);
  #pragma unroll
  for(int i=0;i<8;++i){
    int c = i*256 + tid;
    out[(size_t)r*DI + c] = f2bf(v[i]*rs*w[c]);
  }
}

__global__ void conv_silu(const float* __restrict__ proj, const float* __restrict__ cw,
                          const float* __restrict__ cb, float* __restrict__ xc){
  int idx = blockIdx.x*256 + threadIdx.x;          // over RTOT*DI
  int c = idx & (DI-1); int r = idx >> 11; int t = r & (TT-1);
  float acc = cb[c];
  #pragma unroll
  for(int k=0;k<4;++k){
    int tt = t - 3 + k;
    if(tt >= 0) acc += proj[(size_t)(r-3+k)*LDP + c] * cw[c*4 + k];
  }
  xc[idx] = acc * sigm(acc);
}

__global__ void dt_kernel(const float* __restrict__ proj, const float* __restrict__ Wdt,
                          const float* __restrict__ dt_bias, float* __restrict__ dtb){
  int idx = blockIdx.x*256 + threadIdx.x;          // over RTOT*HS
  int h = idx & 7; int r = idx >> 3;
  float s = dt_bias[h];
  const float* pl = &proj[(size_t)r*LDP + 2*DI];
  #pragma unroll 8
  for(int k=0;k<DTR;++k) s += pl[k]*Wdt[h*DTR + k];
  float sp = (s > 20.f) ? s : log1pf(__expf(s));
  sp = fminf(fmaxf(sp, 1e-4f), 1.0f);
  dtb[idx] = sp;
}

// SSM selective scan. grid = (HS, BSZ), block = 256 (thread = p channel).
__global__ __launch_bounds__(256)
void ssm_scan(const float* __restrict__ proj, const float* __restrict__ xc,
              const float* __restrict__ dtb, const float* __restrict__ A_log,
              const float* __restrict__ Dskip, float* __restrict__ ybuf){
  const int p = threadIdx.x, h = blockIdx.x, b = blockIdx.y;
  __shared__ float sdA[NS], sdtB[NS], sC[NS];
  float aReg = 0.f;
  if(p < NS) aReg = -__expf(A_log[h*NS + p]);
  float hs[NS];
  #pragma unroll
  for(int n=0;n<NS;++n) hs[n] = 0.f;
  const float dsk = Dskip[h*PPC + p];
  for(int t=0;t<TT;++t){
    size_t row = (size_t)(b*TT + t);
    if(p < NS){
      float dtv  = dtb[row*HS + h];
      float bval = proj[row*LDP + 2*DI + DTR + h*NS + p];
      sdtB[p] = dtv*bval;
      sdA[p]  = __expf(dtv*aReg);
    } else if(p < 2*NS){
      sC[p-NS] = proj[row*LDP + 2*DI + DTR + HS*NS + h*NS + (p-NS)];
    }
    __syncthreads();
    float x = xc[row*DI + h*PPC + p];
    float acc = 0.f;
    #pragma unroll
    for(int n=0;n<NS;++n){
      hs[n] = hs[n]*sdA[n] + x*sdtB[n];
      acc += hs[n]*sC[n];
    }
    ybuf[row*DI + h*PPC + p] = acc + dsk*x;
    __syncthreads();
  }
}

__global__ void rope_cast(const float* __restrict__ qf, const float* __restrict__ kf,
                          bfbits* __restrict__ qb, bfbits* __restrict__ kb){
  int idx = blockIdx.x*256 + threadIdx.x;          // over BSZ*TT*HATT*32
  int i = idx & 31, h = (idx>>5) & 15, t = (idx>>9) & (TT-1), b = idx >> 20;
  float fr = __expf(-((float)(2*i)/64.f) * 9.210340372f);   // 10000^(-2i/64)
  float ang = (float)t * fr, s, c;
  __sincosf(ang, &s, &c);
  size_t base = ((size_t)(b*TT + t))*DD + h*64;
  float a0 = qf[base+i], a1 = qf[base+i+32];
  qb[base+i]    = f2bf(a0*c - a1*s);
  qb[base+i+32] = f2bf(a1*c + a0*s);
  float b0 = kf[base+i], b1 = kf[base+i+32];
  kb[base+i]    = f2bf(b0*c - b1*s);
  kb[base+i+32] = f2bf(b1*c + b0*s);
}

// ---------------- sliding-window attention, WMMA for QK^T and P*V ----------------
// grid = (TT/128, HATT, BSZ), block = 256 (8 waves, each wave owns 16 query rows).
#define VTSS 136
#define PSS  136
__global__ __launch_bounds__(256)
void attn_swa(const bfbits* __restrict__ qb, const bfbits* __restrict__ kb,
              const bfbits* __restrict__ vb, float* __restrict__ yout){
  __shared__ __align__(16) bfbits vts[64*VTSS];      // V^T: [d][key]
  __shared__ __align__(16) bfbits ps[8][16*PSS];     // per-wave P tile: [qrow][key]
  const int tid=threadIdx.x, lane=tid&31, wave=tid>>5;
  const int lrow=lane&15, lhi=lane>>4;
  const int t0 = blockIdx.x*128, h = blockIdx.y, b = blockIdx.z;
  const size_t headoff = (size_t)h*64;

  // q fragments for this wave's 16 rows (two K=32 steps over HD=64)
  Frag qf[2];
  {
    size_t qrow = (size_t)(b*TT + t0 + wave*16 + lrow);
    const bfbits* g = qb + qrow*DD + headoff;
    #pragma unroll
    for(int kk=0;kk<2;++kk){
      const bfbits* pp = g + kk*32 + lhi*8;
      qf[kk].q[0] = *(const u32x4*)pp;
      qf[kk].q[1] = *(const u32x4*)(pp + 16);
    }
  }

  float mrow[8], lsum[8];
  #pragma unroll
  for(int v=0;v<8;++v){ mrow[v] = -1e30f; lsum[v] = 0.f; }
  v8f yacc[4];
  #pragma unroll
  for(int ni=0;ni<4;++ni) yacc[ni] = (v8f){};

  for(int kt=0; kt<2; ++kt){
    const int kbase = t0 - 128 + kt*128;

    // stage V^T for this 128-key pass
    {
      int i = tid>>1, dseg = (tid&1)*32;
      int tk = kbase + i;
      if(tk >= 0){
        const bfbits* g = vb + ((size_t)(b*TT + tk))*DD + headoff + dseg;
        #pragma unroll 8
        for(int j=0;j<32;++j) vts[(dseg+j)*VTSS + i] = g[j];
      } else {
        #pragma unroll 8
        for(int j=0;j<32;++j) vts[(dseg+j)*VTSS + i] = 0;
      }
    }
    __syncthreads();

    // scores: 8 key sub-tiles of 16, K=64 in two WMMA steps
    v8f sacc[8];
    #pragma unroll
    for(int kst=0;kst<8;++kst) sacc[kst] = (v8f){};
    #pragma unroll
    for(int kst=0;kst<8;++kst){
      int tk = kbase + kst*16 + lrow;
      int tkc = (tk < 0) ? 0 : tk;
      const bfbits* g = kb + ((size_t)(b*TT + tkc))*DD + headoff;
      #pragma unroll
      for(int kk=0;kk<2;++kk){
        Frag kfrag;
        const bfbits* pp = g + kk*32 + lhi*8;
        kfrag.q[0] = *(const u32x4*)pp;
        kfrag.q[1] = *(const u32x4*)(pp + 16);
        sacc[kst] = __builtin_amdgcn_wmma_f32_16x16x32_bf16(
            false, qf[kk].v, false, kfrag.v, (short)0, sacc[kst], false, false);
      }
    }

    // online softmax over this pass, write P (bf16) into per-wave LDS
    float scale_v[8];
    #pragma unroll
    for(int v=0;v<8;++v){
      int r = lhi*8 + v;                      // query row within wave tile
      float sv8[8];
      float mloc = -1e30f;
      #pragma unroll
      for(int kst=0;kst<8;++kst){
        int kidx = kt*128 + kst*16 + lrow;    // key index within 256 window
        int rel  = wave*16 + r + 128 - kidx;  // t_q - t_k
        int tk   = t0 - 128 + kidx;
        bool ok  = (rel >= 0) && (rel < WINSZ) && (tk >= 0);
        float sv = ok ? sacc[kst][v]*0.125f : -1e30f;   // 1/sqrt(64)
        sv8[kst] = sv;
        mloc = fmaxf(mloc, sv);
      }
      #pragma unroll
      for(int m=1;m<16;m<<=1) mloc = fmaxf(mloc, __shfl_xor(mloc, m));
      float mnew = fmaxf(mrow[v], mloc);
      float sc   = __expf(mrow[v] - mnew);
      float psum = 0.f;
      #pragma unroll
      for(int kst=0;kst<8;++kst){
        float pj = (sv8[kst] > -1e29f) ? __expf(sv8[kst] - mnew) : 0.f;
        psum += pj;
        ps[wave][r*PSS + kst*16 + lrow] = f2bf(pj);
      }
      #pragma unroll
      for(int m=1;m<16;m<<=1) psum += __shfl_xor(psum, m);
      lsum[v] = lsum[v]*sc + psum;
      mrow[v] = mnew;
      scale_v[v] = sc;
    }
    #pragma unroll
    for(int ni=0;ni<4;++ni){
      #pragma unroll
      for(int v=0;v<8;++v) yacc[ni][v] *= scale_v[v];
    }

    // P (16x128) x V (128x64): 4 K-steps, 4 d sub-tiles
    #pragma unroll
    for(int kc=0;kc<4;++kc){
      Frag pa;
      const bfbits* pp = &ps[wave][lrow*PSS + kc*32 + lhi*8];
      pa.q[0] = *(const u32x4*)pp;
      pa.q[1] = *(const u32x4*)(pp + 16);
      #pragma unroll
      for(int ni=0;ni<4;++ni){
        Frag vf;
        const bfbits* pv = &vts[(ni*16 + lrow)*VTSS + kc*32 + lhi*8];
        vf.q[0] = *(const u32x4*)pv;
        vf.q[1] = *(const u32x4*)(pv + 16);
        yacc[ni] = __builtin_amdgcn_wmma_f32_16x16x32_bf16(
            false, pa.v, false, vf.v, (short)0, yacc[ni], false, false);
      }
    }
    __syncthreads();
  }

  #pragma unroll
  for(int ni=0;ni<4;++ni){
    #pragma unroll
    for(int v=0;v<8;++v){
      int r = lhi*8 + v;
      float val = yacc[ni][v] / (lsum[v] + 1e-30f);
      yout[((size_t)(b*TT + t0 + wave*16 + r))*DD + headoff + ni*16 + lrow] = val;
    }
  }
}

// ---------------- host-side orchestration ----------------
static inline size_t alignup(size_t x){ return (x + 255) & ~(size_t)255; }

extern "C" void kernel_launch(void* const* d_in, const int* in_sizes, int n_in,
                              void* d_out, int out_size, void* d_ws, size_t ws_size,
                              hipStream_t stream) {
  (void)in_sizes; (void)n_in; (void)out_size; (void)ws_size;
  const float* x        = (const float*)d_in[0];
  const float* norm1_w  = (const float*)d_in[1];
  const float* norm2_w  = (const float*)d_in[2];
  const float* Win      = (const float*)d_in[3];
  const float* Wdt      = (const float*)d_in[4];
  const float* conv_w   = (const float*)d_in[5];
  const float* conv_b   = (const float*)d_in[6];
  const float* A_log    = (const float*)d_in[7];
  const float* Dskip    = (const float*)d_in[8];
  const float* dt_bias  = (const float*)d_in[9];
  const float* ssd_nw   = (const float*)d_in[10];
  const float* Wout_ssd = (const float*)d_in[11];
  const float* Wres     = (const float*)d_in[12];
  const float* Wq       = (const float*)d_in[13];
  const float* Wk       = (const float*)d_in[14];
  const float* Wv       = (const float*)d_in[15];
  const float* Wo       = (const float*)d_in[16];
  float* out = (float*)d_out;
  char* ws = (char*)d_ws;

  // bump allocator with lifetime-based aliasing
  size_t off = 0;
  #define WSALLOC(name, bytes) char* name = ws + off; off = alignup(off + (size_t)(bytes))
  WSALLOC(winb_,  (size_t)LDP*DD*2);          // padded Win bf16
  WSALLOC(wresb_, (size_t)DI*DD*2);
  WSALLOC(woutb_, (size_t)DD*DI*2);
  WSALLOC(wqb_,   (size_t)DD*DD*2);
  WSALLOC(wkb_,   (size_t)DD*DD*2);
  WSALLOC(wvb_,   (size_t)DD*DD*2);
  WSALLOC(wob_,   (size_t)DD*DD*2);
  WSALLOC(u1b_,   (size_t)RTOT*DD*2);         // also u2b
  WSALLOC(proj_,  (size_t)RTOT*LDP*4);        // also qf/kf/vf (3 x 16MB <= 73MB)
  WSALLOC(xc_,    (size_t)RTOT*DI*4);         // also attnY (16MB <= 32MB)
  WSALLOC(dtb_,   (size_t)RTOT*HS*4);
  WSALLOC(ybuf_,  (size_t)RTOT*DI*4);         // also qb/kb/vb bf16 (24MB <= 32MB)
  WSALLOC(ybf_,   (size_t)RTOT*DI*2);         // also attnYb (8MB <= 16MB)
  WSALLOC(hbuf_,  (size_t)RTOT*DD*4);
  #undef WSALLOC

  bfbits* winb  = (bfbits*)winb_;  bfbits* wresb = (bfbits*)wresb_;
  bfbits* woutb = (bfbits*)woutb_; bfbits* wqb = (bfbits*)wqb_;
  bfbits* wkb   = (bfbits*)wkb_;   bfbits* wvb = (bfbits*)wvb_;
  bfbits* wob   = (bfbits*)wob_;
  bfbits* u1b   = (bfbits*)u1b_;                       // reused as u2b
  float*  proj  = (float*)proj_;
  float*  xc    = (float*)xc_;
  float*  dtb   = (float*)dtb_;
  float*  ybuf  = (float*)ybuf_;
  bfbits* ybf   = (bfbits*)ybf_;
  float*  hbuf  = (float*)hbuf_;
  // aliases
  float*  qf = (float*)proj_;
  float*  kf = (float*)(proj_ + (size_t)RTOT*DD*4);
  float*  vf = (float*)(proj_ + (size_t)RTOT*DD*8);
  bfbits* qbb = (bfbits*)ybuf_;
  bfbits* kbb = (bfbits*)(ybuf_ + (size_t)RTOT*DD*2);
  bfbits* vbb = (bfbits*)(ybuf_ + (size_t)RTOT*DD*4);
  float*  attnY  = (float*)xc_;
  bfbits* attnYb = (bfbits*)ybf_;

  // 1) weight casts (Win padded 4416 -> 4480 rows of zeros)
  cast_pad_bf16<<<(LDP*DD+255)/256,256,0,stream>>>(winb,  Win,      (long)4416*DD, (long)LDP*DD);
  cast_pad_bf16<<<(DI*DD+255)/256, 256,0,stream>>>(wresb, Wres,     (long)DI*DD,   (long)DI*DD);
  cast_pad_bf16<<<(DD*DI+255)/256, 256,0,stream>>>(woutb, Wout_ssd, (long)DD*DI,   (long)DD*DI);
  cast_pad_bf16<<<(DD*DD+255)/256, 256,0,stream>>>(wqb,   Wq,       (long)DD*DD,   (long)DD*DD);
  cast_pad_bf16<<<(DD*DD+255)/256, 256,0,stream>>>(wkb,   Wk,       (long)DD*DD,   (long)DD*DD);
  cast_pad_bf16<<<(DD*DD+255)/256, 256,0,stream>>>(wvb,   Wv,       (long)DD*DD,   (long)DD*DD);
  cast_pad_bf16<<<(DD*DD+255)/256, 256,0,stream>>>(wob,   Wo,       (long)DD*DD,   (long)DD*DD);

  // 2) u1 = rms(x) -> bf16
  rmsnorm_bf16<<<RTOT,256,0,stream>>>(x, norm1_w, u1b, DD);
  // 3) proj = u1 @ Win^T  (N padded to 4480)
  gemm_bf16_wmma<<<dim3(LDP/128, RTOT/128),256,0,stream>>>(u1b, DD, winb, DD, proj, LDP, nullptr, 0, DD);
  // 4) xc = silu(dwconv(xin))
  conv_silu<<<(RTOT*DI)/256,256,0,stream>>>(proj, conv_w, conv_b, xc);
  // 5) dt = clip(softplus(dtl @ Wdt^T + bias))
  dt_kernel<<<(RTOT*HS)/256,256,0,stream>>>(proj, Wdt, dt_bias, dtb);
  // 6) selective scan -> ybuf (includes Dskip*xh)
  ssm_scan<<<dim3(HS,BSZ),256,0,stream>>>(proj, xc, dtb, A_log, Dskip, ybuf);
  // 7) ybuf += u1 @ Wres^T
  gemm_bf16_wmma<<<dim3(DI/128, RTOT/128),256,0,stream>>>(u1b, DD, wresb, DD, ybuf, DI, ybuf, DI, DD);
  // 8) gated RMS -> bf16
  gate_rms_bf16<<<RTOT,256,0,stream>>>(ybuf, proj, ssd_nw, ybf);
  // 9) h = x + y @ Wout^T
  gemm_bf16_wmma<<<dim3(DD/128, RTOT/128),256,0,stream>>>(ybf, DI, woutb, DI, hbuf, DD, x, DD, DI);
  // 10) u2 = rms(h) -> bf16 (reuses u1b)
  rmsnorm_bf16<<<RTOT,256,0,stream>>>(hbuf, norm2_w, u1b, DD);
  // 11) q,k,v GEMMs (into proj space)
  gemm_bf16_wmma<<<dim3(DD/128, RTOT/128),256,0,stream>>>(u1b, DD, wqb, DD, qf, DD, nullptr, 0, DD);
  gemm_bf16_wmma<<<dim3(DD/128, RTOT/128),256,0,stream>>>(u1b, DD, wkb, DD, kf, DD, nullptr, 0, DD);
  gemm_bf16_wmma<<<dim3(DD/128, RTOT/128),256,0,stream>>>(u1b, DD, wvb, DD, vf, DD, nullptr, 0, DD);
  // 12) RoPE(q,k) -> bf16 ; v -> bf16
  rope_cast<<<(RTOT*HATT*32)/256,256,0,stream>>>(qf, kf, qbb, kbb);
  cast_f32_bf16<<<(RTOT*DD)/256,256,0,stream>>>(vbb, vf, (long)RTOT*DD);
  // 13) sliding-window attention -> attnY (fp32)
  attn_swa<<<dim3(TT/128, HATT, BSZ),256,0,stream>>>(qbb, kbb, vbb, attnY);
  // 14) attnY -> bf16
  cast_f32_bf16<<<(RTOT*DD)/256,256,0,stream>>>(attnYb, attnY, (long)RTOT*DD);
  // 15) out = h + attnY @ Wo^T
  gemm_bf16_wmma<<<dim3(DD/128, RTOT/128),256,0,stream>>>(attnYb, DD, wob, DD, out, DD, hbuf, DD, DD);
}
